// EnhancedGlobalGNN_36051955482704
// MI455X (gfx1250) — matrix operand order repaired and checked by reading, not compiled, not run on hardware
//
#include <hip/hip_runtime.h>
#include <math.h>

typedef __attribute__((ext_vector_type(16))) _Float16 v16h;
typedef __attribute__((ext_vector_type(8)))  _Float16 v8h;
typedef __attribute__((ext_vector_type(8)))  float    v8f;

#define kB   8
#define kN   1024
#define kH   4
#define kD   16
#define kHID 64
#define kNC  20
#define kOUT 128
#define kFIN (kN + 4)          // 1028 input features

__device__ __forceinline__ float gelu_exact(float v) {
    return 0.5f * v * (1.0f + erff(v * 0.70710678118654752f));
}

// WMMA f16 A-fragment K offset for VGPR pair v (v in 0..7), lane half q.
__device__ __forceinline__ int wmma_kA(int v, int q) {
    return (v < 4) ? (q * 8 + 2 * v) : (16 + q * 8 + 2 * (v - 4));
}

// ---------------------------------------------------------------------------
// K1: input projection via WMMA (16-node tile x 64 outputs, K=1024 adj cols),
// then deg/coords tail + bias, LN, GELU, community gating in f32 via LDS.
// grid = B*(N/16) blocks of 128 threads (4 waves; wave w -> out cols w*16..)
// ---------------------------------------------------------------------------
__global__ void k_input(const float* __restrict__ adj, const float* __restrict__ coords,
                        const int* __restrict__ comm, const float* __restrict__ in_w,
                        const float* __restrict__ in_b, const float* __restrict__ ln_g,
                        const float* __restrict__ ln_b, const float* __restrict__ ctab,
                        const float* __restrict__ gate_w, const float* __restrict__ gate_b,
                        float* __restrict__ x)
{
    const int blk  = blockIdx.x;
    const int b    = blk / (kN / 16);
    const int n0   = (blk % (kN / 16)) * 16;
    const int w    = threadIdx.x >> 5;     // wave id = output col group
    const int lane = threadIdx.x & 31;
    const int col  = lane & 15;            // A-row node idx / B-col / C-col
    const int q    = lane >> 4;
    const int t    = threadIdx.x;

    __shared__ float stile[16][kHID + 4];  // per-node pre-LN outputs, then post-LN
    __shared__ float sce[16][kHID + 4];    // community embeddings
    __shared__ float sdeg[16];
    __shared__ float scoord[16][3];
    __shared__ float smean[16];
    __shared__ float srstd[16];

    const float* arow = adj + ((size_t)b * kN + n0 + col) * kN;   // A: node row
    const float* wrow = in_w + (w * 16 + col) * kFIN;             // B: in_w row

    v8f c = {};
    float dsum = 0.f;
    for (int kb = 0; kb < kN; kb += 32) {
        if (kb + 32 < kN) __builtin_prefetch(arow + kb + 32, 0, 0);
        v16h a, bf;
#pragma unroll
        for (int v = 0; v < 8; ++v) {
            const int ka = wmma_kA(v, q);
            const float2 xa = *reinterpret_cast<const float2*>(arow + kb + ka);
            dsum += fabsf(xa.x) + fabsf(xa.y);
            a[2 * v]     = (_Float16)xa.x;
            a[2 * v + 1] = (_Float16)xa.y;
            const int kbo = wmma_kA(v, q);
            const float2 wv = *reinterpret_cast<const float2*>(wrow + kb + kbo);
            bf[2 * v]     = (_Float16)wv.x;
            bf[2 * v + 1] = (_Float16)wv.y;
        }
        c = __builtin_amdgcn_wmma_f32_16x16x32_f16(false, a, false, bf, (short)0, c, false, false);
    }

    // |adj| row-sum: lane q=0 covers K{0..7,16..23} of each chunk, q=1 the rest
    dsum += __shfl_xor(dsum, 16, 32);
    if (w == 0 && lane < 16) {
        sdeg[col] = dsum;
        scoord[col][0] = coords[((size_t)b * kN + n0 + col) * 3 + 0];
        scoord[col][1] = coords[((size_t)b * kN + n0 + col) * 3 + 1];
        scoord[col][2] = coords[((size_t)b * kN + n0 + col) * 3 + 2];
    }
    __syncthreads();

    // tail: bias + deg*w[1024] + coords . w[1025..1027]  (per output col jg)
    {
        const int jg = w * 16 + col;
        const float bias = in_b[jg];
        const float4 tw = *reinterpret_cast<const float4*>(in_w + jg * kFIN + kN);
#pragma unroll
        for (int v = 0; v < 8; ++v) {
            const int row = q * 8 + v;
            const float y = c[v] + bias + sdeg[row] * tw.x + scoord[row][0] * tw.y
                          + scoord[row][1] * tw.z + scoord[row][2] * tw.w;
            stile[row][jg] = y;
        }
    }
    __syncthreads();

    // LN stats per node row
    if (t < 16) {
        float s = 0.f;
        for (int k = 0; k < kHID; ++k) s += stile[t][k];
        const float mean = s * (1.0f / (float)kHID);
        float v2 = 0.f;
        for (int k = 0; k < kHID; ++k) { const float d = stile[t][k] - mean; v2 += d * d; }
        smean[t] = mean;
        srstd[t] = rsqrtf(v2 * (1.0f / (float)kHID) + 1e-5f);
    }
    __syncthreads();

    // normalize + GELU; stage community embedding
    {
        const int row = t >> 3;
        const int c0  = (t & 7) * 8;
        int cid = comm[(size_t)b * kN + n0 + row];
        if (cid == -1) cid = kNC;
        cid = (cid < 0) ? 0 : ((cid > kNC) ? kNC : cid);
        const float mean = smean[row], rstd = srstd[row];
#pragma unroll
        for (int k = c0; k < c0 + 8; ++k) {
            const float xv = (stile[row][k] - mean) * rstd * ln_g[k] + ln_b[k];
            stile[row][k] = gelu_exact(xv);
            sce[row][k] = ctab[cid * kHID + k];
        }
    }
    __syncthreads();

    // community gating + final write: thread handles (row, h = (t&7)+8j)
    {
        const int row = t >> 3;
#pragma unroll
        for (int j = 0; j < 8; ++j) {
            const int h = (t & 7) + 8 * j;
            float ga = gate_b[h];
            const float* gw = gate_w + h * (2 * kHID);
            for (int k = 0; k < kHID; ++k)
                ga += stile[row][k] * gw[k] + sce[row][k] * gw[kHID + k];
            const float gs = 1.0f / (1.0f + __expf(-ga));
            x[((size_t)b * kN + n0 + row) * kHID + h] =
                stile[row][h] * (1.0f - gs) + sce[row][h] * gs;
        }
    }
}

// ---------------------------------------------------------------------------
// K2: h = x @ W^T via WMMA f16 (16x16x32); also s = h.a_src, t = h.a_dst,
// and transposed f16 copy ht[b][head][d][n] for the P@V kernel.
// grid = B*(N/16) blocks of 128 threads (wave w -> head w, 16-node tile)
// ---------------------------------------------------------------------------
__global__ void k_gat_h(const float* __restrict__ x, const float* __restrict__ Wl,
                        const float* __restrict__ av, _Float16* __restrict__ ht,
                        float* __restrict__ sbuf, float* __restrict__ tbuf)
{
    const int blk  = blockIdx.x;
    const int b    = blk / (kN / 16);
    const int n0   = (blk % (kN / 16)) * 16;
    const int w    = threadIdx.x >> 5;      // head = wave id
    const int lane = threadIdx.x & 31;
    const int col  = lane & 15;
    const int q    = lane >> 4;

    const float* xrow = x + ((size_t)b * kN + n0 + col) * kHID;
    const float* wrow = Wl + (w * 16 + col) * kHID;

    v8f c = {};
    for (int kb = 0; kb < 2; ++kb) {
        v16h a, bf;
        const int kbase = kb * 32;
#pragma unroll
        for (int v = 0; v < 8; ++v) {
            const int ka = wmma_kA(v, q);
            const float2 xa = *reinterpret_cast<const float2*>(xrow + kbase + ka);
            a[2 * v]     = (_Float16)xa.x;
            a[2 * v + 1] = (_Float16)xa.y;
            const int kbo = wmma_kA(v, q);
            const float2 wv = *reinterpret_cast<const float2*>(wrow + kbase + kbo);
            bf[2 * v]     = (_Float16)wv.x;
            bf[2 * v + 1] = (_Float16)wv.y;
        }
        c = __builtin_amdgcn_wmma_f32_16x16x32_f16(false, a, false, bf, (short)0, c, false, false);
    }

    // ht[b][head][d=col][n]: rows q*8+v contiguous in n -> one 16B store
    {
        v8h hv;
#pragma unroll
        for (int v = 0; v < 8; ++v) hv[v] = (_Float16)c[v];
        _Float16* dst = ht + (((size_t)b * kH + w) * kD + col) * kN + n0 + q * 8;
        *reinterpret_cast<v8h*>(dst) = hv;
    }

    __shared__ float tileh[4][16][17];
#pragma unroll
    for (int v = 0; v < 8; ++v) tileh[w][q * 8 + v][col] = c[v];
    __syncthreads();
    if (lane < 16) {
        const float* as = av + w * (2 * kD);
        float ss = 0.f, tt = 0.f;
        for (int d = 0; d < kD; ++d) {
            const float hv = tileh[w][lane][d];
            ss += hv * as[d];
            tt += hv * as[kD + d];
        }
        const size_t idx = ((size_t)b * kH + w) * kN + n0 + lane;
        sbuf[idx] = ss;
        tbuf[idx] = tt;
    }
}

// ---------------------------------------------------------------------------
// K2b: masked-softmax row stats (max, sum) per (b,head,n). 1 wave per row.
// ---------------------------------------------------------------------------
__global__ void k_stats(const float* __restrict__ adj, const float* __restrict__ sbuf,
                        const float* __restrict__ tbuf, float* __restrict__ rmax,
                        float* __restrict__ rsum)
{
    const int idx  = blockIdx.x;          // (b*H + h)*N + n
    const int n    = idx % kN;
    const int bh   = idx / kN;
    const int b    = bh / kH;
    const int lane = threadIdx.x;

    const float* arow = adj + ((size_t)b * kN + n) * kN;
    const float* trow = tbuf + (size_t)bh * kN;
    const float  sn   = sbuf[(size_t)bh * kN + n];

    float mmax = -INFINITY, msum = 0.f;
    for (int m = lane; m < kN; m += 32) {
        const float am = arow[m];
        if (am > 0.f) {
            float e = sn + trow[m];
            e = (e > 0.f) ? e : 0.2f * e;
            if (e > mmax) { msum = msum * __expf(mmax - e) + 1.0f; mmax = e; }
            else          { msum += __expf(e - mmax); }
        }
    }
#pragma unroll
    for (int off = 16; off > 0; off >>= 1) {
        const float om = __shfl_xor(mmax, off, 32);
        const float os = __shfl_xor(msum, off, 32);
        const float nm = fmaxf(mmax, om);
        float su = 0.f;
        if (mmax > -3e38f) su += msum * __expf(mmax - nm);
        if (om   > -3e38f) su += os   * __expf(om - nm);
        mmax = nm; msum = su;
    }
    if (lane == 0) { rmax[idx] = mmax; rsum[idx] = msum; }
}

// ---------------------------------------------------------------------------
// K3: o = softmax(e) @ h via WMMA f16; probabilities regenerated in-register
// (flash-style, pre-computed row stats). grid = B*(N/16), 128 thr (wave=head).
// ---------------------------------------------------------------------------
__global__ void k_gat_agg(const float* __restrict__ adj, const float* __restrict__ sbuf,
                          const float* __restrict__ tbuf, const float* __restrict__ rmax,
                          const float* __restrict__ rsum, const _Float16* __restrict__ ht,
                          float* __restrict__ obuf)
{
    const int blk  = blockIdx.x;
    const int b    = blk / (kN / 16);
    const int n0   = (blk % (kN / 16)) * 16;
    const int w    = threadIdx.x >> 5;
    const int lane = threadIdx.x & 31;
    const int col  = lane & 15;
    const int q    = lane >> 4;

    const size_t bh   = (size_t)b * kH + w;
    const int    nA   = n0 + col;
    const float  sn   = sbuf[bh * kN + nA];
    const float  rmx  = rmax[bh * kN + nA];
    const float* arow = adj + ((size_t)b * kN + nA) * kN;
    const float* trow = tbuf + bh * kN;
    const _Float16* hrow = ht + (bh * kD + col) * kN;

    v8f c = {};
    for (int mb = 0; mb < kN; mb += 32) {
        if (mb + 32 < kN) __builtin_prefetch(arow + mb + 32, 0, 0);
        v16h a, bf;
#pragma unroll
        for (int v = 0; v < 8; ++v) {
            const int ka = q * 16 + 2 * v;
            const int m  = mb + ka;
            const float2 am = *reinterpret_cast<const float2*>(arow + m);
            const float2 tm = *reinterpret_cast<const float2*>(trow + m);
            float e0 = sn + tm.x; e0 = (e0 > 0.f) ? e0 : 0.2f * e0;
            float e1 = sn + tm.y; e1 = (e1 > 0.f) ? e1 : 0.2f * e1;
            const float p0 = (am.x > 0.f) ? __expf(e0 - rmx) : 0.f;
            const float p1 = (am.y > 0.f) ? __expf(e1 - rmx) : 0.f;
            a[2 * v]     = (_Float16)p0;
            a[2 * v + 1] = (_Float16)p1;
        }
        const v8h* bp = reinterpret_cast<const v8h*>(hrow + mb + q * 16);
        const v8h blo = bp[0], bhi = bp[1];
#pragma unroll
        for (int i = 0; i < 8; ++i) { bf[i] = blo[i]; bf[8 + i] = bhi[i]; }
        c = __builtin_amdgcn_wmma_f32_16x16x32_f16(false, a, false, bf, (short)0, c, false, false);
    }
#pragma unroll
    for (int v = 0; v < 8; ++v) {
        const int row = q * 8 + v;
        const int n   = n0 + row;
        const float rs  = rsum[bh * kN + n];
        const float inv = (rs > 0.f) ? (1.0f / rs) : 0.f;
        obuf[((size_t)b * kN + n) * kHID + w * kD + col] = c[v] * inv;
    }
}

// ---------------------------------------------------------------------------
// K4: x = LN(o + x)   (grid B*N, 64 threads)
// ---------------------------------------------------------------------------
__global__ void k_ln_res(const float* __restrict__ obuf, float* __restrict__ x,
                         const float* __restrict__ g, const float* __restrict__ bta)
{
    const int bn = blockIdx.x;
    const int t  = threadIdx.x;
    __shared__ float sred[64];
    const float v = obuf[(size_t)bn * kHID + t] + x[(size_t)bn * kHID + t];
    sred[t] = v; __syncthreads();
    for (int s = 32; s > 0; s >>= 1) { if (t < s) sred[t] += sred[t + s]; __syncthreads(); }
    const float mean = sred[0] * (1.0f / 64.0f); __syncthreads();
    const float d = v - mean;
    sred[t] = d * d; __syncthreads();
    for (int s = 32; s > 0; s >>= 1) { if (t < s) sred[t] += sred[t + s]; __syncthreads(); }
    const float var = sred[0] * (1.0f / 64.0f); __syncthreads();
    x[(size_t)bn * kHID + t] = d * rsqrtf(var + 1e-5f) * g[t] + bta[t];
}

// ---------------------------------------------------------------------------
// K5: pooling (mean/max/attention) + output head (Linear + LN + GELU)
// grid = B blocks of 256 threads
// ---------------------------------------------------------------------------
__global__ void k_pool_out(const float* __restrict__ x, const float* __restrict__ pw,
                           const float* __restrict__ pb, const float* __restrict__ ow,
                           const float* __restrict__ ob, const float* __restrict__ og,
                           const float* __restrict__ olb, float* __restrict__ out)
{
    const int b = blockIdx.x;
    const int t = threadIdx.x;           // 0..255
    __shared__ float sred[256];
    __shared__ float psum[4][64];
    __shared__ float pmax[4][64];
    __shared__ float pattn[4][64];
    __shared__ float sw[kN];
    __shared__ float sg[3 * kHID];

    const float* xb = x + (size_t)b * kN * kHID;
    const int f  = t & 63;
    const int ch = t >> 6;

    float acc = 0.f, mx = -INFINITY;
    for (int n = ch * 256; n < ch * 256 + 256; ++n) {
        const float v = xb[(size_t)n * kHID + f];
        acc += v; mx = fmaxf(mx, v);
    }
    psum[ch][f] = acc; pmax[ch][f] = mx;

    float sc[4]; float lmax = -INFINITY;
    for (int j = 0; j < 4; ++j) {
        const int n = t + 256 * j;
        float d = pb[0];
        const float* xr = xb + (size_t)n * kHID;
        for (int k = 0; k < kHID; ++k) d += xr[k] * pw[k];
        sc[j] = tanhf(d);
        lmax = fmaxf(lmax, sc[j]);
    }
    __syncthreads();
    sred[t] = lmax; __syncthreads();
    for (int s = 128; s > 0; s >>= 1) { if (t < s) sred[t] = fmaxf(sred[t], sred[t + s]); __syncthreads(); }
    const float smx = sred[0]; __syncthreads();
    float lsum = 0.f; float pe[4];
    for (int j = 0; j < 4; ++j) { pe[j] = __expf(sc[j] - smx); lsum += pe[j]; }
    sred[t] = lsum; __syncthreads();
    for (int s = 128; s > 0; s >>= 1) { if (t < s) sred[t] += sred[t + s]; __syncthreads(); }
    const float ssum = sred[0]; __syncthreads();
    for (int j = 0; j < 4; ++j) sw[t + 256 * j] = pe[j] / ssum;
    __syncthreads();

    float aacc = 0.f;
    for (int n = ch * 256; n < ch * 256 + 256; ++n) aacc += xb[(size_t)n * kHID + f] * sw[n];
    pattn[ch][f] = aacc;
    __syncthreads();

    if (t < 3 * kHID) {
        float val;
        if (t < 64)       val = (psum[0][t] + psum[1][t] + psum[2][t] + psum[3][t]) * (1.0f / (float)kN);
        else if (t < 128) { const int ff = t - 64;  val = fmaxf(fmaxf(pmax[0][ff], pmax[1][ff]), fmaxf(pmax[2][ff], pmax[3][ff])); }
        else              { const int ff = t - 128; val = pattn[0][ff] + pattn[1][ff] + pattn[2][ff] + pattn[3][ff]; }
        sg[t] = val;
    }
    __syncthreads();

    float o = 0.f;
    if (t < kOUT) {
        o = ob[t];
        const float* wr = ow + t * (3 * kHID);
        for (int j = 0; j < 3 * kHID; ++j) o += sg[j] * wr[j];
    }
    sred[t] = (t < kOUT) ? o : 0.f; __syncthreads();
    for (int s = 128; s > 0; s >>= 1) { if (t < s) sred[t] += sred[t + s]; __syncthreads(); }
    const float mean = sred[0] * (1.0f / (float)kOUT); __syncthreads();
    const float d2 = o - mean;
    sred[t] = (t < kOUT) ? d2 * d2 : 0.f; __syncthreads();
    for (int s = 128; s > 0; s >>= 1) { if (t < s) sred[t] += sred[t + s]; __syncthreads(); }
    const float var = sred[0] * (1.0f / (float)kOUT); __syncthreads();
    if (t < kOUT) {
        float y = d2 * rsqrtf(var + 1e-5f) * og[t] + olb[t];
        out[b * kOUT + t] = gelu_exact(y);
    }
}

// ---------------------------------------------------------------------------
extern "C" void kernel_launch(void* const* d_in, const int* in_sizes, int n_in,
                              void* d_out, int out_size, void* d_ws, size_t ws_size,
                              hipStream_t stream)
{
    (void)in_sizes; (void)n_in; (void)out_size; (void)ws_size;
    const float* adj      = (const float*)d_in[0];
    const float* coords   = (const float*)d_in[1];
    const int*   comm     = (const int*)  d_in[2];
    const float* in_w     = (const float*)d_in[3];
    const float* in_b     = (const float*)d_in[4];
    const float* in_ln_g  = (const float*)d_in[5];
    const float* in_ln_b  = (const float*)d_in[6];
    const float* ctab     = (const float*)d_in[7];
    const float* gate_w   = (const float*)d_in[8];
    const float* gate_b   = (const float*)d_in[9];
    const float* gat_W    = (const float*)d_in[10];
    const float* gat_a    = (const float*)d_in[11];
    const float* gat_ln_g = (const float*)d_in[12];
    const float* gat_ln_b = (const float*)d_in[13];
    const float* pool_w   = (const float*)d_in[14];
    const float* pool_b   = (const float*)d_in[15];
    const float* out_w    = (const float*)d_in[16];
    const float* out_b    = (const float*)d_in[17];
    const float* out_ln_g = (const float*)d_in[18];
    const float* out_ln_b = (const float*)d_in[19];

    float* xbuf  = (float*)d_ws;
    float* obuf  = xbuf  + (size_t)kB * kN * kHID;
    float* sbuf  = obuf  + (size_t)kB * kN * kHID;
    float* tbuf  = sbuf  + (size_t)kB * kH * kN;
    float* rmaxb = tbuf  + (size_t)kB * kH * kN;
    float* rsumb = rmaxb + (size_t)kB * kH * kN;
    _Float16* htb = (_Float16*)(rsumb + (size_t)kB * kH * kN);

    k_input<<<kB * (kN / 16), 128, 0, stream>>>(adj, coords, comm, in_w, in_b, in_ln_g,
                                                in_ln_b, ctab, gate_w, gate_b, xbuf);

    for (int l = 0; l < 3; ++l) {
        k_gat_h<<<kB * (kN / 16), 128, 0, stream>>>(xbuf, gat_W + l * kHID * kHID,
                                                    gat_a + l * kH * 2 * kD, htb, sbuf, tbuf);
        k_stats<<<kB * kH * kN, 32, 0, stream>>>(adj, sbuf, tbuf, rmaxb, rsumb);
        k_gat_agg<<<kB * (kN / 16), 128, 0, stream>>>(adj, sbuf, tbuf, rmaxb, rsumb, htb, obuf);
        k_ln_res<<<kB * kN, 64, 0, stream>>>(obuf, xbuf, gat_ln_g + l * kHID, gat_ln_b + l * kHID);
    }

    k_pool_out<<<kB, 256, 0, stream>>>(xbuf, pool_w, pool_b, out_w, out_b,
                                       out_ln_g, out_ln_b, (float*)d_out);
}
